// FiLMLSTMCell_12867722019349
// MI455X (gfx1250) — matrix operand order repaired and checked by reading, not compiled, not run on hardware
//
#include <hip/hip_runtime.h>
#include <hip/hip_bf16.h>

typedef __attribute__((ext_vector_type(16))) __bf16 v16bf;
typedef __attribute__((ext_vector_type(8)))  __bf16 v8bf;
typedef __attribute__((ext_vector_type(8)))  float  v8f;

// clang vector type (not HIP_vector_type struct) for the async-LDS builtin
typedef int vi4 __attribute__((vector_size(16)));
typedef __attribute__((address_space(1))) vi4* gvi4p;   // global
typedef __attribute__((address_space(3))) vi4* lvi4p;   // LDS

#if __has_builtin(__builtin_amdgcn_global_load_async_to_lds_b128)
#define HAVE_ASYNC_LDS 1
#else
#define HAVE_ASYNC_LDS 0
#endif

union FragBF {
  v16bf v;
  v8bf  h[2];
};

__device__ __forceinline__ float sigf(float x) { return 1.0f / (1.0f + __expf(-x)); }

#if HAVE_ASYNC_LDS
__device__ __forceinline__ void wait_async_8() {
#if __has_builtin(__builtin_amdgcn_s_wait_asynccnt)
  __builtin_amdgcn_s_wait_asynccnt(8);
#else
  asm volatile("s_wait_asynccnt 0x8" ::: "memory");
#endif
}
__device__ __forceinline__ void wait_async_0() {
#if __has_builtin(__builtin_amdgcn_s_wait_asynccnt)
  __builtin_amdgcn_s_wait_asynccnt(0);
#else
  asm volatile("s_wait_asynccnt 0x0" ::: "memory");
#endif
}
#endif

// ---------------------------------------------------------------------------
// Kernel P: W [1024][2048] f32 (K-major)  ->  Wt [2048][1024] bf16 (N-major)
// 32x32 tiles via LDS so both global read and write are coalesced.
// ---------------------------------------------------------------------------
__global__ void __launch_bounds__(256)
wconv_kernel(const float* __restrict__ W, __bf16* __restrict__ Wt) {
  __shared__ float tile[32][33];
  const int kb = blockIdx.x * 32;          // K base (1024/32 = 32 blocks)
  const int nb = blockIdx.y * 32;          // N base (2048/32 = 64 blocks)
  const int tx = threadIdx.x & 31;
  const int ty = threadIdx.x >> 5;         // 0..7
#pragma unroll
  for (int i = 0; i < 32; i += 8)
    tile[ty + i][tx] = W[(size_t)(kb + ty + i) * 2048 + nb + tx];
  __syncthreads();
#pragma unroll
  for (int i = 0; i < 32; i += 8)
    Wt[(size_t)(nb + ty + i) * 1024 + kb + tx] = (__bf16)tile[tx][ty + i];
}

// ---------------------------------------------------------------------------
// Kernel G: fused GEMM (bf16 WMMA, fp32 acc) + bias + GroupNorm(512) + FiLM.
// Grid: (B/32, 4 gates). Block: 256 threads = 8 waves.
// Block tile: 32 rows x 512 cols (one GN group == one LSTM gate block).
// W panel is double-buffered and streamed memory->LDS with
// GLOBAL_LOAD_ASYNC_TO_LDS_B128 (ASYNCcnt), overlapping WMMA of chunk k with
// the fetch of chunk k+1. Falls back to sync staging if builtin absent.
// ---------------------------------------------------------------------------
__global__ void __launch_bounds__(256)
gemm_gn_film_kernel(const float* __restrict__ x, const float* __restrict__ h,
                    const __bf16* __restrict__ Wt, const float* __restrict__ b_lin,
                    const float* __restrict__ u, float* __restrict__ zbuf) {
#if HAVE_ASYNC_LDS
  __shared__ __align__(16) __bf16 Wlds[2][512][40];  // double-buffered W panel
#else
  __shared__ __align__(16) __bf16 Wlds[1][512][40];
#endif
  __shared__ __align__(16) __bf16 Alds[32][40];      // 32 rows x 32 K (+8 pad)
  __shared__ float zlds[32][520];                    // post-GEMM tile (padded)
  __shared__ float red[32][16];                      // per-row partial sums
  __shared__ float rowstat[32][2];                   // mean, rstd

  const int tid   = threadIdx.x;
  const int lane  = tid & 31;
  const int wave  = tid >> 5;
  const int half  = lane >> 4;                       // 0 or 1
  const int r     = lane & 15;
  const int mtile = wave & 1;
  const int ngrp  = wave >> 1;                       // 0..3
  const int rowBase = blockIdx.x * 32;
  const int gate  = blockIdx.y;                      // 0..3

  v8f acc[8] = {};

  const int ar = tid >> 3;                           // A-chunk row 0..31
  const int ac = (tid & 7) * 4;                      // A-chunk col (float4)

#if HAVE_ASYNC_LDS
  // Issue one 512x32 bf16 W panel (2048 x 16B) as 8 async b128 loads/thread.
  auto issueW = [&](int kkN, int buf) {
#pragma unroll
    for (int i = 0; i < 8; ++i) {
      const int q   = i * 256 + tid;
      const int n   = q >> 2;
      const int seg = q & 3;
      gvi4p gsrc = (gvi4p)(Wt + (size_t)(gate * 512 + n) * 1024 + kkN) + seg;
      lvi4p ldst = (lvi4p)&Wlds[buf][n][seg * 8];
      __builtin_amdgcn_global_load_async_to_lds_b128(gsrc, ldst, 0, 0);
    }
  };
  issueW(0, 0);  // prologue: chunk 0 -> buffer 0
#endif

  const int kSteps = 32;
  for (int step = 0; step < kSteps; ++step) {
    const int kk = step * 32;
    // --- stage A chunk: rows [rowBase, +32), K cols [kk, kk+32) of cat(x,h)
    {
      const int k = kk + ac;                         // uniform x/h split per kk
      const float* src = (k < 512) ? (x + (size_t)(rowBase + ar) * 512 + k)
                                   : (h + (size_t)(rowBase + ar) * 512 + (k - 512));
      const float4 f = *(const float4*)src;
      Alds[ar][ac + 0] = (__bf16)f.x;
      Alds[ar][ac + 1] = (__bf16)f.y;
      Alds[ar][ac + 2] = (__bf16)f.z;
      Alds[ar][ac + 3] = (__bf16)f.w;
    }
#if HAVE_ASYNC_LDS
    // --- prefetch next W panel async, then wait for current panel (in-order)
    if (step + 1 < kSteps) {
      issueW(kk + 32, (step + 1) & 1);
      wait_async_8();   // newest 8 outstanding -> current chunk complete
    } else {
      wait_async_0();
    }
    const int wbuf = step & 1;
#else
    // --- sync staging of W panel
#pragma unroll
    for (int i = 0; i < 8; ++i) {
      const int q   = i * 256 + tid;
      const int n   = q >> 2;
      const int seg = q & 3;
      const vi4* gsrc = (const vi4*)(Wt + (size_t)(gate * 512 + n) * 1024 + kk);
      *(vi4*)&Wlds[0][n][seg * 8] = gsrc[seg];
    }
    const int wbuf = 0;
#endif
    __syncthreads();

    // --- A fragment (16x32 bf16): half 0 -> K{0..7,16..23}, half 1 -> K{8..15,24..31}
    FragBF fa;
    fa.h[0] = *(const v8bf*)&Alds[mtile * 16 + r][half * 8];
    fa.h[1] = *(const v8bf*)&Alds[mtile * 16 + r][16 + half * 8];
#pragma unroll
    for (int nt = 0; nt < 8; ++nt) {
      // --- B fragment (32x16 bf16): half 0 -> K{0..7,16..23}, half 1 -> K{8..15,24..31}
      const int n = (ngrp * 8 + nt) * 16 + r;
      FragBF fb;
      fb.h[0] = *(const v8bf*)&Wlds[wbuf][n][half * 8];
      fb.h[1] = *(const v8bf*)&Wlds[wbuf][n][16 + half * 8];
      acc[nt] = __builtin_amdgcn_wmma_f32_16x16x32_bf16(
          false, fa.v, false, fb.v, (short)0, acc[nt], false, false);
    }
    __syncthreads();
  }

  // --- epilogue: bias + spill tile to LDS (C layout: M = v + 8*half, N = r)
#pragma unroll
  for (int nt = 0; nt < 8; ++nt) {
    const int col  = (ngrp * 8 + nt) * 16 + r;
    const float bias = b_lin[gate * 512 + col];
#pragma unroll
    for (int v = 0; v < 8; ++v)
      zlds[mtile * 16 + half * 8 + v][col] = acc[nt][v] + bias;
  }
  __syncthreads();

  // --- GroupNorm over the 512 cols of each row (8 threads per row)
  const int rr = tid >> 3;
  const int l8 = tid & 7;
  {
    float s = 0.f, s2 = 0.f;
    for (int c = l8; c < 512; c += 8) {
      const float z = zlds[rr][c];
      s += z; s2 += z * z;
    }
    red[rr][l8]     = s;
    red[rr][8 + l8] = s2;
  }
  __syncthreads();
  if (l8 == 0) {
    float S = 0.f, S2 = 0.f;
#pragma unroll
    for (int j = 0; j < 8; ++j) { S += red[rr][j]; S2 += red[rr][8 + j]; }
    const float mean = S * (1.0f / 512.0f);
    const float var  = S2 * (1.0f / 512.0f) - mean * mean;
    rowstat[rr][0] = mean;
    rowstat[rr][1] = rsqrtf(var + 1e-5f);
  }
  __syncthreads();

  // --- normalize + FiLM, write to z workspace (coalesced)
  for (int idx = tid; idx < 32 * 512; idx += 256) {
    const int rw   = idx >> 9;
    const int c    = idx & 511;
    const int rowg = rowBase + rw;
    const int colg = gate * 512 + c;
    const float zn = (zlds[rw][c] - rowstat[rw][0]) * rowstat[rw][1];
    const float fa_ = u[(size_t)rowg * 4096 + colg];           // a = u[:, :2048]
    const float fb_ = u[(size_t)rowg * 4096 + 2048 + colg];    // b = u[:, 2048:]
    zbuf[(size_t)rowg * 2048 + colg] = zn * (1.0f + fa_) + fb_;
  }
}

// ---------------------------------------------------------------------------
// Kernel L: LSTM gate math + output GroupNorm(1 group, affine) per row.
// Grid: B blocks of 512 threads (one thread per channel).
// out = [h_new (B*512) | c_new (B*512)]
// ---------------------------------------------------------------------------
__global__ void __launch_bounds__(512)
lstm_out_kernel(const float* __restrict__ zbuf, const float* __restrict__ cprev,
                const float* __restrict__ gamma, const float* __restrict__ beta,
                float* __restrict__ out, int B) {
  __shared__ float ps[16], ps2[16], stat[2];
  const int row = blockIdx.x;
  const int t   = threadIdx.x;
  const float* zr = zbuf + (size_t)row * 2048;
  const float zi = zr[t];
  const float zf = zr[512 + t];
  const float zo = zr[1024 + t];
  const float zg = zr[1536 + t];
  const float cp = cprev[(size_t)row * 512 + t];
  const float cn = sigf(zf) * cp + sigf(zi) * tanhf(zg);

  float s = cn, s2 = cn * cn;
#pragma unroll
  for (int off = 16; off > 0; off >>= 1) {
    s  += __shfl_down(s,  off, 32);
    s2 += __shfl_down(s2, off, 32);
  }
  const int lane = t & 31, wv = t >> 5;
  if (lane == 0) { ps[wv] = s; ps2[wv] = s2; }
  __syncthreads();
  if (t == 0) {
    float S = 0.f, S2 = 0.f;
#pragma unroll
    for (int j = 0; j < 16; ++j) { S += ps[j]; S2 += ps2[j]; }
    const float mean = S * (1.0f / 512.0f);
    const float var  = S2 * (1.0f / 512.0f) - mean * mean;
    stat[0] = mean;
    stat[1] = rsqrtf(var + 1e-5f);
  }
  __syncthreads();
  const float cnorm = (cn - stat[0]) * stat[1] * gamma[t] + beta[t];
  const float hn = sigf(zo) * tanhf(cnorm);
  out[(size_t)row * 512 + t] = hn;
  out[(size_t)B * 512 + (size_t)row * 512 + t] = cn;
}

// ---------------------------------------------------------------------------
extern "C" void kernel_launch(void* const* d_in, const int* in_sizes, int n_in,
                              void* d_out, int out_size, void* d_ws, size_t ws_size,
                              hipStream_t stream) {
  const float* x     = (const float*)d_in[0];
  const float* h     = (const float*)d_in[1];
  const float* c     = (const float*)d_in[2];
  const float* u     = (const float*)d_in[3];
  const float* W     = (const float*)d_in[4];
  const float* b_lin = (const float*)d_in[5];
  const float* gamma = (const float*)d_in[6];
  const float* beta  = (const float*)d_in[7];
  float* out = (float*)d_out;
  const int B = in_sizes[0] / 512;   // 16384

  // workspace: Wt bf16 [2048*1024] (4 MB) | zbuf f32 [B*2048] (128 MB)
  __bf16* Wt  = (__bf16*)d_ws;
  float* zbuf = (float*)((char*)d_ws + (size_t)2048 * 1024 * sizeof(__bf16));

  wconv_kernel<<<dim3(32, 64), 256, 0, stream>>>(W, Wt);
  gemm_gn_film_kernel<<<dim3(B / 32, 4), 256, 0, stream>>>(x, h, Wt, b_lin, u, zbuf);
  lstm_out_kernel<<<B, 512, 0, stream>>>(zbuf, c, gamma, beta, out, B);
}
